// GraphLayerTransformerLinear_51101520888318
// MI455X (gfx1250) — matrix operand order repaired and checked by reading, not compiled, not run on hardware
//
#include <hip/hip_runtime.h>
#include <math.h>

// ---------------------------------------------------------------------------
// MI455X / gfx1250 implementation.
// All heavy math runs through v_wmma_f32_16x16x32_bf16 (f32 accumulate).
// Generic batched GEMM: 128x128x64 block tile, 256 threads = 8 waves (wave32),
// each wave owns a 64x32 sub-tile = 4x2 WMMA accumulators, 2 k-substeps.
// A/B tiles stream global->LDS with GLOBAL_LOAD_ASYNC_TO_LDS_B128 (ASYNCcnt),
// double-buffered: tile k+1 loads overlap WMMA on tile k.
// ---------------------------------------------------------------------------

typedef __attribute__((ext_vector_type(16))) __bf16        v16bf;
typedef __attribute__((ext_vector_type(8)))  float         v8f;
typedef __attribute__((ext_vector_type(4)))  unsigned int  uint4v;

__device__ __forceinline__ unsigned short f32_to_bf16(float f) {
    unsigned int u = __builtin_bit_cast(unsigned int, f);
    u += 0x7FFFu + ((u >> 16) & 1u);          // round-to-nearest-even
    return (unsigned short)(u >> 16);
}

union Frag16 { uint4v u[2]; v16bf v; };       // 32 bytes: one bf16 fragment slice

// ---------------------------------------------------------------------------
// Generic batched bf16 GEMM with fused epilogue.
//   C[z] = A[z] * B[z] + bias ; act==1 -> softplus ; writes f32 and/or bf16.
// Batch offset model: off = (z / bi) * sHi + (z % bi) * sLo   (elements)
// M = gridDim.y*128, N = gridDim.x*128; K multiple of 64.
// ---------------------------------------------------------------------------
__global__ __launch_bounds__(256)
void gemm_bf16_kernel(const unsigned short* __restrict__ A, long long sAhi, long long sAlo, int lda,
                      const unsigned short* __restrict__ B, long long sBhi, long long sBlo, int ldb,
                      const float* __restrict__ bias,
                      float* __restrict__ Cf, unsigned short* __restrict__ Cbf,
                      long long sChi, long long sClo, int ldc,
                      int K, int bi, int act)
{
    constexpr int BM = 128, BN = 128, BK = 64;
    __shared__ unsigned short As[2][BM * BK]; // 2 x 16 KB  [m][k]
    __shared__ unsigned short Bs[2][BK * BN]; // 2 x 16 KB  [k][n]

    const int z  = blockIdx.z;
    const int zo = z / bi, zi = z % bi;
    A += (size_t)(zo * sAhi + zi * sAlo);
    B += (size_t)(zo * sBhi + zi * sBlo);
    const long long offC = zo * sChi + zi * sClo;
    if (Cf)  Cf  += offC;
    if (Cbf) Cbf += offC;

    const int bm = blockIdx.y * BM;
    const int bn = blockIdx.x * BN;
    const int tid  = threadIdx.x;
    const int wid  = tid >> 5;
    const int lane = tid & 31;
    const int wm = (wid >> 2) * 64;           // wave origin inside block tile
    const int wn = (wid & 3) * 32;

    // LDS byte offsets of the staging buffers (generic addr low 32b = LDS offset)
    const unsigned asBase = (unsigned)(unsigned long long)(const void*)&As[0][0];
    const unsigned bsBase = (unsigned)(unsigned long long)(const void*)&Bs[0][0];

    v8f acc[4][2];
    const v8f vzero = {0.f, 0.f, 0.f, 0.f, 0.f, 0.f, 0.f, 0.f};
#pragma unroll
    for (int i = 0; i < 4; ++i)
#pragma unroll
        for (int j = 0; j < 2; ++j) acc[i][j] = vzero;

    const int mloc  = lane & 15;
    const int khalf = lane >> 4;

    // Async-copy one 128x64 A tile + 64x128 B tile into LDS buffer `buf`.
    auto issueTile = [&](int k0, int buf) {
        const unsigned aB = asBase + (unsigned)buf * (BM * BK * 2);
        const unsigned bB = bsBase + (unsigned)buf * (BK * BN * 2);
#pragma unroll
        for (int i = 0; i < 4; ++i) {
            const int c = tid + 256 * i;                  // 1024 chunks of 8 bf16
            {   // A tile: 8 chunks per row
                const int r = c >> 3, cc = (c & 7) * 8;
                const unsigned long long g =
                    (unsigned long long)(A + (size_t)(bm + r) * lda + k0 + cc);
                const unsigned l = aB + (unsigned)((r * BK + cc) * 2);
                asm volatile("global_load_async_to_lds_b128 %0, %1, off"
                             :: "v"(l), "v"(g) : "memory");
            }
            {   // B tile: 16 chunks per row
                const int r = c >> 4, cc = (c & 15) * 8;
                const unsigned long long g =
                    (unsigned long long)(B + (size_t)(k0 + r) * ldb + bn + cc);
                const unsigned l = bB + (unsigned)((r * BN + cc) * 2);
                asm volatile("global_load_async_to_lds_b128 %0, %1, off"
                             :: "v"(l), "v"(g) : "memory");
            }
        }
    };

    const int nk = K / BK;
    issueTile(0, 0);
    for (int kt = 0; kt < nk; ++kt) {
        const int cur = kt & 1;
        asm volatile("s_wait_asynccnt 0x0" ::: "memory"); // own tile landed in LDS
        __syncthreads();                                  // publish to all waves
        if (kt + 1 < nk) issueTile((kt + 1) * BK, cur ^ 1); // overlap next tile

#pragma unroll
        for (int ks = 0; ks < 2; ++ks) {                  // two 32-wide k-substeps
            Frag16 af[4], bfr[2];
#pragma unroll
            for (int mf = 0; mf < 4; ++mf) {
                const unsigned short* src =
                    &As[cur][(wm + mf * 16 + mloc) * BK + ks * 32];
                af[mf].u[0] = *(const uint4v*)(src + khalf * 8);      // K 0..7 / 8..15
                af[mf].u[1] = *(const uint4v*)(src + 16 + khalf * 8); // K 16..23 / 24..31
            }
#pragma unroll
            for (int nf = 0; nf < 2; ++nf) {
                const unsigned short* src =
                    &Bs[cur][(ks * 32 + khalf * 8) * BN + wn + nf * 16 + mloc];
#pragma unroll
                for (int s = 0; s < 16; ++s)                          // K 0..7 / 8..15, then K 16..23 / 24..31
                    bfr[nf].v[s] = __builtin_bit_cast(__bf16, src[(s < 8 ? s : s + 8) * BN]);
            }
#pragma unroll
            for (int mf = 0; mf < 4; ++mf)
#pragma unroll
                for (int nf = 0; nf < 2; ++nf)
                    acc[mf][nf] = __builtin_amdgcn_wmma_f32_16x16x32_bf16(
                        false, af[mf].v, false, bfr[nf].v,
                        (short)0, acc[mf][nf], false, false);
        }
    }

    // ---- epilogue: bias, optional softplus, f32/bf16 stores ----
    const int col  = lane & 15;
    const int hsel = lane >> 4;
#pragma unroll
    for (int mf = 0; mf < 4; ++mf) {
#pragma unroll
        for (int nf = 0; nf < 2; ++nf) {
            const int gn = bn + wn + nf * 16 + col;
            const float bv = bias ? bias[gn] : 0.0f;
            const int gm0 = bm + wm + mf * 16 + hsel * 8;
#pragma unroll
            for (int j = 0; j < 8; ++j) {
                float v = acc[mf][nf][j] + bv;
                if (act == 1) v = (v > 20.0f) ? v : log1pf(__expf(v));
                const size_t idx = (size_t)(gm0 + j) * ldc + gn;
                if (Cf)  Cf[idx]  = v;
                if (Cbf) Cbf[idx] = f32_to_bf16(v);
            }
        }
    }
}

// ---------------------------------------------------------------------------
// Elementwise / helper kernels
// ---------------------------------------------------------------------------
__global__ void conv_bf16_kernel(const float* __restrict__ in, unsigned short* __restrict__ out, int n) {
    int i = blockIdx.x * blockDim.x + threadIdx.x;
    if (i < n) out[i] = f32_to_bf16(in[i]);
}

// lin_W [513,512] -> padded bf16 [576,512] (zero rows 513..575)
__global__ void pad_wlin_kernel(const float* __restrict__ W, unsigned short* __restrict__ out) {
    int i = blockIdx.x * blockDim.x + threadIdx.x;
    if (i < 576 * 512) {
        int r = i >> 9;
        out[i] = (r < 513) ? f32_to_bf16(W[i]) : (unsigned short)0;
    }
}

// A for ConcatLinear: [2048,576] = [t | x | zero-pad]
__global__ void build_ain_kernel(const float* __restrict__ x, const float* __restrict__ t,
                                 unsigned short* __restrict__ out) {
    int i = blockIdx.x * blockDim.x + threadIdx.x;
    if (i < 2048 * 576) {
        int tok = i / 576, c = i - tok * 576;
        float v = (c == 0) ? t[0] : (c <= 512 ? x[(size_t)tok * 512 + (c - 1)] : 0.0f);
        out[i] = f32_to_bf16(v);
    }
}

// k_bf [ (n*256+key), h*512+d ] -> kt [ nh ][ d ][ key ]; LDS-tiled 32x32, both
// global sides coalesced. grid = (d/32, key/32, nh)
__global__ __launch_bounds__(256)
void transpose_k_kernel(const unsigned short* __restrict__ kin,
                        unsigned short* __restrict__ kt) {
    __shared__ unsigned short tile[32][33];
    const int nh = blockIdx.z, n = nh >> 3, h = nh & 7;
    const int d0 = blockIdx.x * 32, key0 = blockIdx.y * 32;
    const int c = threadIdx.x & 31;
    const int r = threadIdx.x >> 5;            // 0..7
#pragma unroll
    for (int i = 0; i < 4; ++i) {
        const int krow = r + i * 8;
        tile[krow][c] = kin[(size_t)(n * 256 + key0 + krow) * 4096 + h * 512 + d0 + c];
    }
    __syncthreads();
#pragma unroll
    for (int i = 0; i < 4; ++i) {
        const int drow = r + i * 8;
        kt[((size_t)nh * 512 + d0 + drow) * 256 + key0 + c] = tile[c][drow];
    }
}

// masked softmax over rows of scores[nh][q][k]; P bf16. One wave per row.
__global__ __launch_bounds__(256)
void softmax_kernel(const float* __restrict__ scores, const int* __restrict__ masks,
                    unsigned short* __restrict__ P) {
    const int wid = threadIdx.x >> 5, lane = threadIdx.x & 31;
    const int row = blockIdx.x * 8 + wid;       // < 16384
    const int nh = row >> 8, q = row & 255, n = nh >> 3;
    const int mq = masks[n * 256 + q];
    const float* s = scores + (size_t)row * 256;
    float vals[8]; int mk[8];
    float mx = -3.0e38f;
#pragma unroll
    for (int i = 0; i < 8; ++i) {
        int k = lane + i * 32;
        mk[i] = masks[n * 256 + k];
        float v = (mq && mk[i]) ? s[k] : -1.0e30f;
        vals[i] = v; mx = fmaxf(mx, v);
    }
    for (int o = 16; o > 0; o >>= 1) mx = fmaxf(mx, __shfl_xor(mx, o, 32));
    float sum = 0.f;
#pragma unroll
    for (int i = 0; i < 8; ++i) {
        float e = (mq && mk[i]) ? __expf(vals[i] - mx) : 0.f;
        vals[i] = e; sum += e;
    }
    for (int o = 16; o > 0; o >>= 1) sum += __shfl_xor(sum, o, 32);
    const float inv = (sum > 0.f) ? 1.f / sum : 0.f;
#pragma unroll
    for (int i = 0; i < 8; ++i)
        P[(size_t)row * 256 + lane + i * 32] = f32_to_bf16(vals[i] * inv);
}

// y = LayerNorm(fc + h) * g + b, masked; bf16 out. One block per token.
__global__ __launch_bounds__(256)
void ln_kernel(const float* __restrict__ fc, const float* __restrict__ h,
               const float* __restrict__ g, const float* __restrict__ b,
               const int* __restrict__ masks, unsigned short* __restrict__ out) {
    __shared__ float red[256];
    const int tok = blockIdx.x, tid = threadIdx.x;
    const float* fr = fc + (size_t)tok * 512;
    const float* hr = h  + (size_t)tok * 512;
    const float a0 = fr[tid] + hr[tid];
    const float a1 = fr[tid + 256] + hr[tid + 256];
    red[tid] = a0 + a1; __syncthreads();
    for (int s = 128; s > 0; s >>= 1) { if (tid < s) red[tid] += red[tid + s]; __syncthreads(); }
    const float mean = red[0] * (1.f / 512.f); __syncthreads();
    const float d0 = a0 - mean, d1 = a1 - mean;
    red[tid] = d0 * d0 + d1 * d1; __syncthreads();
    for (int s = 128; s > 0; s >>= 1) { if (tid < s) red[tid] += red[tid + s]; __syncthreads(); }
    const float rs = rsqrtf(red[0] * (1.f / 512.f) + 1e-5f);
    const float mf = masks[tok] ? 1.f : 0.f;
    out[(size_t)tok * 512 + tid]       = f32_to_bf16((d0 * rs * g[tid]       + b[tid])       * mf);
    out[(size_t)tok * 512 + tid + 256] = f32_to_bf16((d1 * rs * g[tid + 256] + b[tid + 256]) * mf);
}

// d_out = [ h*mask  |  mlpU + mlpB ]
__global__ void pack_kernel(const float* __restrict__ h, const float* __restrict__ mu,
                            const float* __restrict__ mb, const int* __restrict__ masks,
                            float* __restrict__ out) {
    int i = blockIdx.x * blockDim.x + threadIdx.x;
    if (i < 2048 * 512) {
        float mf = masks[i >> 9] ? 1.f : 0.f;
        out[i] = h[i] * mf;
        out[1048576 + i] = mu[i] + mb[i];
    }
}

// ---------------------------------------------------------------------------
// Host orchestration
// ---------------------------------------------------------------------------
extern "C" void kernel_launch(void* const* d_in, const int* in_sizes, int n_in,
                              void* d_out, int out_size, void* d_ws, size_t ws_size,
                              hipStream_t stream) {
    (void)in_sizes; (void)n_in; (void)out_size; (void)ws_size;
    // Input map (setup_inputs dict order, params flattened by insertion order):
    // 0:x 1:t 2:masks 3:lin_W 4:lin_b
    // 5..14 unary  : wq_W wq_b wk_W wk_b wv_W wv_b fc_W fc_b ln_g ln_b
    // 15..24 binary: same
    // 25..34 mlp   : W0 b0 W1 b1 W2 b2 W3 b3 W4 b4
    const float* x     = (const float*)d_in[0];
    const float* t     = (const float*)d_in[1];
    const int*   masks = (const int*)d_in[2];
    float* out = (float*)d_out;

    constexpr int MTOK = 2048, D = 512, HD = 4096, KP = 576, HID = 2048;

    char* w = (char*)d_ws;
    size_t off = 0;
    auto alloc = [&](size_t bytes) -> void* {
        void* p = w + off; off = (off + bytes + 255) & ~(size_t)255; return p;
    };
    unsigned short* wlin = (unsigned short*)alloc((size_t)KP * D * 2);
    unsigned short* wq   = (unsigned short*)alloc((size_t)D * HD * 2);
    unsigned short* wk   = (unsigned short*)alloc((size_t)D * HD * 2);
    unsigned short* wv   = (unsigned short*)alloc((size_t)D * HD * 2);
    unsigned short* wfc  = (unsigned short*)alloc((size_t)HD * D * 2);
    unsigned short* mw0  = (unsigned short*)alloc((size_t)D * HID * 2);
    unsigned short* mw1  = (unsigned short*)alloc((size_t)HID * HID * 2);
    unsigned short* mw2  = (unsigned short*)alloc((size_t)HID * HID * 2);
    unsigned short* mw3  = (unsigned short*)alloc((size_t)HID * HID * 2);
    unsigned short* mw4  = (unsigned short*)alloc((size_t)HID * D * 2);
    unsigned short* ain  = (unsigned short*)alloc((size_t)MTOK * KP * 2);
    float*          hf   = (float*)alloc((size_t)MTOK * D * 4);
    unsigned short* hb   = (unsigned short*)alloc((size_t)MTOK * D * 2);
    unsigned short* qb   = (unsigned short*)alloc((size_t)MTOK * HD * 2);
    unsigned short* kb   = (unsigned short*)alloc((size_t)MTOK * HD * 2);
    unsigned short* vb   = (unsigned short*)alloc((size_t)MTOK * HD * 2);
    unsigned short* ktb  = (unsigned short*)alloc((size_t)64 * 512 * 256 * 2);
    float*          sc   = (float*)alloc((size_t)64 * 65536 * 4);
    unsigned short* pb   = (unsigned short*)alloc((size_t)64 * 65536 * 2);
    unsigned short* atb  = (unsigned short*)alloc((size_t)MTOK * HD * 2);
    float*          fcf  = (float*)alloc((size_t)MTOK * D * 4);
    unsigned short* brb  = (unsigned short*)alloc((size_t)MTOK * D * 2);
    unsigned short* t1   = (unsigned short*)alloc((size_t)MTOK * HID * 2);
    unsigned short* t2   = (unsigned short*)alloc((size_t)MTOK * HID * 2);
    float* mlpo[2] = { (float*)alloc((size_t)MTOK * D * 4),
                       (float*)alloc((size_t)MTOK * D * 4) };

    auto conv = [&](int idx, unsigned short* dst, int n) {
        conv_bf16_kernel<<<dim3((n + 255) / 256), dim3(256), 0, stream>>>((const float*)d_in[idx], dst, n);
    };
    auto gemm = [&](const unsigned short* A, long long sAhi, long long sAlo, int lda,
                    const unsigned short* B, long long sBhi, long long sBlo, int ldb,
                    const float* bias, float* Cf, unsigned short* Cbf,
                    long long sChi, long long sClo, int ldc,
                    int Mm, int Nn, int Kk, int batches, int bi, int act) {
        dim3 grid(Nn / 128, Mm / 128, batches), block(256);
        gemm_bf16_kernel<<<grid, block, 0, stream>>>(A, sAhi, sAlo, lda, B, sBhi, sBlo, ldb,
                                                     bias, Cf, Cbf, sChi, sClo, ldc, Kk, bi, act);
    };

    // ---- weight prep (every call; no cross-call caching allowed) ----
    pad_wlin_kernel<<<dim3((KP * D + 255) / 256), dim3(256), 0, stream>>>((const float*)d_in[3], wlin);
    conv(25, mw0, D * HID); conv(27, mw1, HID * HID); conv(29, mw2, HID * HID);
    conv(31, mw3, HID * HID); conv(33, mw4, HID * D);
    build_ain_kernel<<<dim3((MTOK * KP + 255) / 256), dim3(256), 0, stream>>>(x, t, ain);

    // ---- h = ConcatLinear(t, x) ----
    gemm(ain, 0, 0, KP, wlin, 0, 0, D, (const float*)d_in[4],
         hf, hb, 0, 0, D, MTOK, D, KP, 1, 1, 0);

    for (int br = 0; br < 2; ++br) {
        const int base = 5 + br * 10;
        conv(base + 0, wq, D * HD); conv(base + 2, wk, D * HD);
        conv(base + 4, wv, D * HD); conv(base + 6, wfc, HD * D);

        // QKV projections
        gemm(hb, 0, 0, D, wq, 0, 0, HD, (const float*)d_in[base + 1], nullptr, qb, 0, 0, HD, MTOK, HD, D, 1, 1, 0);
        gemm(hb, 0, 0, D, wk, 0, 0, HD, (const float*)d_in[base + 3], nullptr, kb, 0, 0, HD, MTOK, HD, D, 1, 1, 0);
        gemm(hb, 0, 0, D, wv, 0, 0, HD, (const float*)d_in[base + 5], nullptr, vb, 0, 0, HD, MTOK, HD, D, 1, 1, 0);

        // K transpose for Q·K^T
        transpose_k_kernel<<<dim3(16, 8, 64), dim3(256), 0, stream>>>(kb, ktb);

        // scores[nh] = Q[n,:,h,:] · Kt[nh]   (batch z = n*8+h, bi=8)
        gemm(qb, 1048576, 512, HD, ktb, 8LL * 131072, 131072, 256, nullptr,
             sc, nullptr, 8LL * 65536, 65536, 256, 256, 256, 512, 64, 8, 0);

        softmax_kernel<<<dim3(2048), dim3(256), 0, stream>>>(sc, masks, pb);

        // attn_out[nh] = P[nh] · V[n,:,h,:]
        gemm(pb, 8LL * 65536, 65536, 256, vb, 1048576, 512, HD, nullptr,
             nullptr, atb, 1048576, 512, HD, 256, 512, 256, 64, 8, 0);

        // fc projection + residual LN + node mask
        gemm(atb, 0, 0, HD, wfc, 0, 0, D, (const float*)d_in[base + 7],
             fcf, nullptr, 0, 0, D, MTOK, D, HD, 1, 1, 0);
        ln_kernel<<<dim3(MTOK), dim3(256), 0, stream>>>(fcf, hf,
            (const float*)d_in[base + 8], (const float*)d_in[base + 9], masks, brb);

        // MLP: Linear -> 3x(Softplus->Linear) -> Linear   (softplus fused in producer epilogue)
        gemm(brb, 0, 0, D,   mw0, 0, 0, HID, (const float*)d_in[26], nullptr, t1, 0, 0, HID, MTOK, HID, D,   1, 1, 1);
        gemm(t1,  0, 0, HID, mw1, 0, 0, HID, (const float*)d_in[28], nullptr, t2, 0, 0, HID, MTOK, HID, HID, 1, 1, 1);
        gemm(t2,  0, 0, HID, mw2, 0, 0, HID, (const float*)d_in[30], nullptr, t1, 0, 0, HID, MTOK, HID, HID, 1, 1, 1);
        gemm(t1,  0, 0, HID, mw3, 0, 0, HID, (const float*)d_in[32], nullptr, t2, 0, 0, HID, MTOK, HID, HID, 1, 1, 0);
        gemm(t2,  0, 0, HID, mw4, 0, 0, D,   (const float*)d_in[34], mlpo[br], nullptr, 0, 0, D, MTOK, D, HID, 1, 1, 0);
    }

    pack_kernel<<<dim3((MTOK * D + 255) / 256), dim3(256), 0, stream>>>(hf, mlpo[0], mlpo[1], masks, out);
}
